// LSTMModel_11836929868302
// MI455X (gfx1250) — matrix operand order repaired and checked by reading, not compiled, not run on hardware
//
#include <hip/hip_runtime.h>
#include <cstdint>
#include <cstddef>

typedef _Float16 half_t;
typedef __attribute__((ext_vector_type(16))) _Float16 v16h;
typedef __attribute__((ext_vector_type(8)))  _Float16 v8h;
typedef __attribute__((ext_vector_type(8)))  float    v8f;
typedef __attribute__((ext_vector_type(4)))  float    f4;
typedef __attribute__((ext_vector_type(4)))  unsigned int u32x4;
typedef __attribute__((ext_vector_type(8)))  int      i32x8;
typedef __attribute__((ext_vector_type(4)))  int      i32x4;

#define SEQ   2048
#define BATCH 64
#define HID   256
#define G4    1024          // 4*H
#define MROWS (SEQ*BATCH)   // 131072
#define GXSTEP 65536        // halves per timestep block of GX (64*1024)

#if defined(__HIP_DEVICE_COMPILE__) && defined(__has_builtin)
#if __has_builtin(__builtin_amdgcn_tensor_load_to_lds) && __has_builtin(__builtin_amdgcn_s_wait_tensorcnt)
#define HAVE_TDM 1
#endif
#endif
#ifndef HAVE_TDM
#define HAVE_TDM 0
#endif

__device__ __forceinline__ v8f wmma32(v16h a, v16h b, v8f c) {
  // v_wmma_f32_16x16x32_f16
  return __builtin_amdgcn_wmma_f32_16x16x32_f16(false, a, false, b, (short)0, c,
                                                false, false);
}

__device__ __forceinline__ float sigf(float x) {
  return 1.f / (1.f + __expf(-x));
}
__device__ __forceinline__ float tanhfast(float x) {
  return 1.f - 2.f / (__expf(2.f * x) + 1.f);   // saturates correctly at +-1
}

// ---------------------------------------------------------------------------
// TDM: DMA one contiguous 128KB GX timestep slab (16384 x 8B) into LDS.
// Descriptor per CDNA5 ISA ch.8: group0 = {count=1 | lds_addr | global_addr |
// type=2}, group1 = {data_size=3(8B), tensor_dim0=tile_dim0=16384,
// tensor_dim1=tile_dim1=1}.
// ---------------------------------------------------------------------------
__device__ __forceinline__ void tdm_issue(const half_t* gsrc, void* ldst) {
#if HAVE_TDM
  unsigned long long ga = (unsigned long long)(uintptr_t)gsrc;
  unsigned int lds = (unsigned int)(uintptr_t)ldst;   // low 32b = LDS offset
  u32x4 g0;
  g0[0] = 1u;                                   // count=1, is_restore=0
  g0[1] = lds;                                  // lds_addr (bytes)
  g0[2] = (unsigned int)(ga & 0xffffffffu);     // global_addr[31:0]
  g0[3] = (unsigned int)((ga >> 32) & 0x01ffffffu) | (2u << 30);  // type=2
  i32x8 g1;
  g1[0] = (int)(3u << 16);          // data_size=3 (8B), no mask/pad/iterate
  g1[1] = (int)(16384u << 16);      // tensor_dim0[15:0] = 16384
  g1[2] = (int)(1u << 16);          // tensor_dim0[31:16]=0, tensor_dim1=1
  g1[3] = (int)(16384u << 16);      // tile_dim0 = 16384
  g1[4] = 1;                        // tile_dim1 = 1, tile_dim2 = 0
  g1[5] = 16384;                    // tensor_dim0_stride
  g1[6] = 0;
  g1[7] = 0;
  i32x4 z4 = {0, 0, 0, 0};
#if __clang_major__ >= 23
  i32x8 z8 = {0, 0, 0, 0, 0, 0, 0, 0};
  __builtin_amdgcn_tensor_load_to_lds(g0, g1, z4, z4, z8, 0);
#else
  __builtin_amdgcn_tensor_load_to_lds(g0, g1, z4, z4, 0);
#endif
#else
  (void)gsrc; (void)ldst;
#endif
}

// ---------------------------------------------------------------------------
// Pack Bmat = [W_ih; W_hh]^T (512 x 1024) into per-WMMA-fragment f16 layout:
// tile (kt,nt) of 32x16; lane l holds 16 halves: K = kt*32 + (l>>4)*8 + e + (e>=8 ? 8 : 0),
// N = nt*16 + (l&15).  One lane-fragment = 32 contiguous bytes.
// ---------------------------------------------------------------------------
__global__ __launch_bounds__(256) void pack_weights(
    const float* __restrict__ Wih0, const float* __restrict__ Whh0,
    const float* __restrict__ Wih1, const float* __restrict__ Whh1,
    half_t* __restrict__ pB0, half_t* __restrict__ pB1)
{
  int tid   = blockIdx.x * 256 + threadIdx.x;   // 0..65535
  int layer = tid >> 15;
  int rem   = tid & 32767;
  int kt    = rem >> 11;          // 0..15
  int nt    = (rem >> 5) & 63;    // 0..63
  int l     = rem & 31;
  const float* Wih = layer ? Wih1 : Wih0;
  const float* Whh = layer ? Whh1 : Whh0;
  half_t* dst = (layer ? pB1 : pB0) + (((size_t)(kt * 64 + nt) * 32 + l) * 16);
  int N = nt * 16 + (l & 15);
  int kb = kt * 32 + ((l & 16) >> 1);
  #pragma unroll
  for (int e = 0; e < 16; ++e) {
    int K = kb + e + ((e >> 3) << 3);
    float v = (K < 256) ? Wih[N * 256 + K] : Whh[N * 256 + (K - 256)];
    dst[e] = (half_t)v;
  }
}

__global__ __launch_bounds__(256) void bias_combine(
    const float* __restrict__ a0, const float* __restrict__ b0,
    const float* __restrict__ a1, const float* __restrict__ b1,
    float* __restrict__ o0, float* __restrict__ o1)
{
  int n = blockIdx.x * 256 + threadIdx.x;
  if (n < 1024)       o0[n] = a0[n] + b0[n];
  else if (n < 2048) { int j = n - 1024; o1[j] = a1[j] + b1[j]; }
}

// ---------------------------------------------------------------------------
// Parallel input projection: GX = (In o maskI) @ W_ih^T + bias, stored f16 in
// accumulator-fragment order: GX[((rtile*64+nt)*32 + lane)*8 + v].
// Block: 256 threads (8 waves), tile 128 rows x 64 cols, K=256 (kt 0..7 of pB).
// ---------------------------------------------------------------------------
__global__ __launch_bounds__(256) void input_gemm(
    const float* __restrict__ In, const float* __restrict__ maskI,
    const half_t* __restrict__ pB, const float* __restrict__ bias,
    half_t* __restrict__ GX)
{
  __shared__ half_t sA[128 * 40];          // 32-K slab, padded stride
  const int tid = threadIdx.x;
  const int l   = tid & 31;
  const int w   = tid >> 5;                // 0..7
  const int a   = w >> 1;                  // row-tile pair 0..3
  const int cb  = w & 1;
  const size_t rowBase = (size_t)blockIdx.x * 128;
  const int ntBase = blockIdx.y * 4 + cb * 2;

  v8f acc[2][2];
  #pragma unroll
  for (int rt = 0; rt < 2; ++rt)
    #pragma unroll
    for (int c = 0; c < 2; ++c) {
      float bv = bias[(ntBase + c) * 16 + (l & 15)];
      #pragma unroll
      for (int i = 0; i < 8; ++i) acc[rt][c][i] = bv;
    }

  const int ldRow = tid >> 1;              // 0..127
  const int ldSeg = (tid & 1) * 16;        // 0 / 16
  const int mrow  = (int)((rowBase + ldRow) & 63);
  const int kh    = (l >> 4) * 8;          // lane-half K offset

  for (int kt = 0; kt < 8; ++kt) {
    { // stage A slab: (In o mask) -> f16 LDS
      const float* src = In    + (rowBase + ldRow) * 256 + kt * 32 + ldSeg;
      const float* msk = maskI + (size_t)mrow * 256      + kt * 32 + ldSeg;
      #pragma unroll
      for (int q = 0; q < 2; ++q) {
        f4 x0 = *(const f4*)(src + q * 8 + 0);
        f4 x1 = *(const f4*)(src + q * 8 + 4);
        f4 m0 = *(const f4*)(msk + q * 8 + 0);
        f4 m1 = *(const f4*)(msk + q * 8 + 4);
        v8h hh;
        #pragma unroll
        for (int e = 0; e < 4; ++e) {
          hh[e]     = (half_t)(x0[e] * m0[e]);
          hh[e + 4] = (half_t)(x1[e] * m1[e]);
        }
        *(v8h*)&sA[ldRow * 40 + ldSeg + q * 8] = hh;
      }
    }
    __syncthreads();

    union { v16h v; v8h h[2]; } A0, A1, B0, B1;
    {
      int row0 = (a * 2 + 0) * 16 + (l & 15);
      int row1 = (a * 2 + 1) * 16 + (l & 15);
      A0.h[0] = *(const v8h*)&sA[row0 * 40 + kh];
      A0.h[1] = *(const v8h*)&sA[row0 * 40 + 16 + kh];
      A1.h[0] = *(const v8h*)&sA[row1 * 40 + kh];
      A1.h[1] = *(const v8h*)&sA[row1 * 40 + 16 + kh];
    }
    B0.v = *(const v16h*)(pB + (((size_t)(kt * 64 + ntBase + 0)) * 32 + l) * 16);
    B1.v = *(const v16h*)(pB + (((size_t)(kt * 64 + ntBase + 1)) * 32 + l) * 16);

    acc[0][0] = wmma32(A0.v, B0.v, acc[0][0]);
    acc[0][1] = wmma32(A0.v, B1.v, acc[0][1]);
    acc[1][0] = wmma32(A1.v, B0.v, acc[1][0]);
    acc[1][1] = wmma32(A1.v, B1.v, acc[1][1]);
    __syncthreads();
  }

  #pragma unroll
  for (int rt = 0; rt < 2; ++rt)
    #pragma unroll
    for (int c = 0; c < 2; ++c) {
      size_t rtile = (size_t)blockIdx.x * 8 + a * 2 + rt;
      size_t off = ((rtile * 64 + (size_t)(ntBase + c)) * 32 + l) * 8;
      v8h o;
      #pragma unroll
      for (int i = 0; i < 8; ++i) o[i] = (half_t)acc[rt][c][i];
      *(v8h*)(GX + off) = o;
    }
}

// ---------------------------------------------------------------------------
// Persistent recurrence: single workgroup, 32 waves on one WGP.
// Wave w: tcol = w&15 (column tile of H), r2 = w>>4 (batch half).
// Each wave owns i/f/g/o for its 32 batch rows x 16 hidden cols -> c,h update
// entirely in registers.  h o mask_h is written back to LDS as f16 for the
// next step's A matrix.  W_hh streamed from L2 in packed fragment order.
// USEGX=1: gate pre-activations (GX) double-buffered in LDS via TDM
// (tensor_load_to_lds + s_wait_tensorcnt), K=256 per step.
// USEGX=0: fused input GEMM per step, K=512.
// ---------------------------------------------------------------------------
template <int USEGX>
__global__ __launch_bounds__(1024, 1) void lstm_rec(
    const float* __restrict__ Xin,   // fused mode only
    const half_t* __restrict__ GX,   // split mode only
    const half_t* __restrict__ pB,
    const float* __restrict__ bias,  // fused mode only
    const float* __restrict__ maskI, // fused mode only
    const float* __restrict__ maskH,
    float* __restrict__ outH, float* __restrict__ outC)
{
  // split: h-only A buffer (stride 272) + 2x128KB GX double buffer (~297KB)
  // fused: full [x|h] A buffer (stride 544, ~70KB)
  constexpr int ASTR  = USEGX ? 272 : 544;
  constexpr int HCOL0 = USEGX ? 0   : 256;   // column base of h in sA
  __shared__ half_t sA[64 * ASTR];
  __shared__ half_t sGX[USEGX ? 2 * GXSTEP : 64];

  const int tid   = threadIdx.x;
  const int l     = tid & 31;
  const int w     = tid >> 5;        // 0..31
  const int tcol  = w & 15;
  const int r2    = w >> 4;          // 0..1
  const int kh    = (l >> 4) * 8;
  const int rowHi = (l >> 4) * 8;    // accumulator M offset for lanes 16..31
  const int jcol  = tcol * 16 + (l & 15);

  for (int i = tid; i < 64 * ASTR; i += 1024) sA[i] = (half_t)0.f;

  // preload variational mask on h for the owned (b, j) positions
  float mh[2][8];
  #pragma unroll
  for (int rt2 = 0; rt2 < 2; ++rt2)
    #pragma unroll
    for (int v = 0; v < 8; ++v) {
      int b = (r2 * 2 + rt2) * 16 + v + rowHi;
      mh[rt2][v] = maskH[b * 256 + jcol];
    }

  v8f cst[2];
  #pragma unroll
  for (int rt2 = 0; rt2 < 2; ++rt2)
    #pragma unroll
    for (int i = 0; i < 8; ++i) cst[rt2][i] = 0.f;

  int buf = 0;
  if (USEGX) {
#if HAVE_TDM
    if (w == 0) tdm_issue(GX, &sGX[0]);          // preload GX(t=0) -> buf 0
#else
    for (int i = tid; i < GXSTEP / 8; i += 1024)
      *(v8h*)&sGX[i * 8] = *(const v8h*)(GX + i * 8);
#endif
  }

  const int ktStart = USEGX ? 8 : 0;   // packed-B k-tiles: 8..15 = W_hh part

  for (int t = 0; t < SEQ; ++t) {
    if (!USEGX) {
      // stage x_t o mask_i into sA[:, 0:256]
      int row = tid >> 4;
      int c0  = (tid & 15) * 16;
      const float* src = Xin   + ((size_t)t * 64 + row) * 256 + c0;
      const float* msk = maskI + (size_t)row * 256 + c0;
      #pragma unroll
      for (int q = 0; q < 2; ++q) {
        f4 x0 = *(const f4*)(src + q * 8 + 0);
        f4 x1 = *(const f4*)(src + q * 8 + 4);
        f4 m0 = *(const f4*)(msk + q * 8 + 0);
        f4 m1 = *(const f4*)(msk + q * 8 + 4);
        v8h hh;
        #pragma unroll
        for (int e = 0; e < 4; ++e) {
          hh[e]     = (half_t)(x0[e] * m0[e]);
          hh[e + 4] = (half_t)(x1[e] * m1[e]);
        }
        *(v8h*)&sA[row * 544 + c0 + q * 8] = hh;
      }
    }
#if HAVE_TDM
    if (USEGX && w == 0) __builtin_amdgcn_s_wait_tensorcnt(0);
#endif
    __syncthreads();   // GX(t) arrived, x staged, prev-step h visible

    if (USEGX && t + 1 < SEQ) {
      // overlap: DMA next step's GX slab into the other LDS buffer
#if HAVE_TDM
      if (w == 0) tdm_issue(GX + (size_t)(t + 1) * GXSTEP, &sGX[(buf ^ 1) * GXSTEP]);
#else
      const half_t* src = GX + (size_t)(t + 1) * GXSTEP;
      half_t* dst = &sGX[(buf ^ 1) * GXSTEP];
      for (int i = tid; i < GXSTEP / 8; i += 1024)
        *(v8h*)&dst[i * 8] = *(const v8h*)(src + i * 8);
#endif
    }

    v8f acc[2][4];
    if (USEGX) {
      const half_t* gb = &sGX[buf * GXSTEP];
      #pragma unroll
      for (int rt2 = 0; rt2 < 2; ++rt2)
        #pragma unroll
        for (int g = 0; g < 4; ++g) {
          int rtloc = r2 * 2 + rt2;            // 0..3 within this step
          int nt = g * 16 + tcol;
          v8h x = *(const v8h*)&gb[((rtloc * 64 + nt) * 32 + l) * 8];
          #pragma unroll
          for (int i = 0; i < 8; ++i) acc[rt2][g][i] = (float)x[i];
        }
    } else {
      #pragma unroll
      for (int g = 0; g < 4; ++g) {
        int nt = g * 16 + tcol;
        float bv = bias[nt * 16 + (l & 15)];
        #pragma unroll
        for (int rt2 = 0; rt2 < 2; ++rt2)
          #pragma unroll
          for (int i = 0; i < 8; ++i) acc[rt2][g][i] = bv;
      }
    }

    for (int kt = ktStart; kt < 16; ++kt) {
      union { v16h v; v8h h[2]; } A0, A1, B;
      {
        int row0 = (r2 * 2 + 0) * 16 + (l & 15);
        int row1 = (r2 * 2 + 1) * 16 + (l & 15);
        int kb = (kt - ktStart) * 32 + kh;     // LDS column of this k-slab
        A0.h[0] = *(const v8h*)&sA[row0 * ASTR + kb];
        A0.h[1] = *(const v8h*)&sA[row0 * ASTR + kb + 16];
        A1.h[0] = *(const v8h*)&sA[row1 * ASTR + kb];
        A1.h[1] = *(const v8h*)&sA[row1 * ASTR + kb + 16];
      }
      if (kt + 1 < 16) {  // prefetch next k-tile of W into L0/L2
        __builtin_prefetch(pB + (((size_t)((kt + 1) * 64 + tcol)) * 32 + l) * 16, 0, 3);
      }
      #pragma unroll
      for (int g = 0; g < 4; ++g) {
        int nt = g * 16 + tcol;
        B.v = *(const v16h*)(pB + (((size_t)(kt * 64 + nt)) * 32 + l) * 16);
        acc[0][g] = wmma32(A0.v, B.v, acc[0][g]);
        acc[1][g] = wmma32(A1.v, B.v, acc[1][g]);
      }
    }
    __syncthreads();   // all reads of sA done before h overwrite

    #pragma unroll
    for (int rt2 = 0; rt2 < 2; ++rt2) {
      #pragma unroll
      for (int i = 0; i < 8; ++i) {
        float gi = acc[rt2][0][i];
        float gf = acc[rt2][1][i];
        float gg = acc[rt2][2][i];
        float go = acc[rt2][3][i];
        float cn = sigf(gf) * cst[rt2][i] + sigf(gi) * tanhfast(gg);
        float hn = sigf(go) * tanhfast(cn);
        cst[rt2][i] = cn;
        int b = (r2 * 2 + rt2) * 16 + i + rowHi;
        size_t o = ((size_t)t * 64 + b) * 256 + jcol;
        outH[o] = hn;
        outC[o] = cn;
        sA[b * ASTR + HCOL0 + jcol] = (half_t)(hn * mh[rt2][i]);
      }
    }
    buf ^= 1;
  }
}

// ---------------------------------------------------------------------------
extern "C" void kernel_launch(void* const* d_in, const int* in_sizes, int n_in,
                              void* d_out, int out_size, void* d_ws, size_t ws_size,
                              hipStream_t stream)
{
  (void)in_sizes; (void)n_in; (void)out_size;
  const float* X    = (const float*)d_in[0];
  const float* Wih0 = (const float*)d_in[1];
  const float* Whh0 = (const float*)d_in[2];
  const float* bih0 = (const float*)d_in[3];
  const float* bhh0 = (const float*)d_in[4];
  const float* Wih1 = (const float*)d_in[5];
  const float* Whh1 = (const float*)d_in[6];
  const float* bih1 = (const float*)d_in[7];
  const float* bhh1 = (const float*)d_in[8];
  const float* mi0  = (const float*)d_in[9];
  const float* mh0  = (const float*)d_in[10];
  const float* mi1  = (const float*)d_in[11];
  const float* mh1  = (const float*)d_in[12];

  char* ws = (char*)d_ws;
  half_t* pB0   = (half_t*)(ws);
  half_t* pB1   = (half_t*)(ws + (size_t)(1u << 20));
  float*  bias0 = (float*) (ws + (size_t)(2u << 20));
  float*  bias1 = (float*) (ws + (size_t)(2u << 20) + 4096);
  half_t* GX    = (half_t*)(ws + (size_t)(2u << 20) + 8192);
  const size_t need = (size_t)(2u << 20) + 8192 + (size_t)MROWS * G4 * sizeof(half_t);
  const int split = (ws_size >= need) ? 1 : 0;

  float* ht1 = (float*)d_out;
  const size_t N1 = (size_t)SEQ * BATCH * HID;
  float* ct1 = ht1 + N1;
  float* ht0 = ht1 + 2 * N1;
  float* ct0 = ht1 + 3 * N1;

  pack_weights<<<256, 256, 0, stream>>>(Wih0, Whh0, Wih1, Whh1, pB0, pB1);
  bias_combine<<<8, 256, 0, stream>>>(bih0, bhh0, bih1, bhh1, bias0, bias1);

  if (split) {
    input_gemm<<<dim3(1024, 16), 256, 0, stream>>>(X, mi0, pB0, bias0, GX);
    lstm_rec<1><<<1, 1024, 0, stream>>>(nullptr, GX, pB0, bias0, nullptr, mh0,
                                        ht0, ct0);
    input_gemm<<<dim3(1024, 16), 256, 0, stream>>>(ht0, mi1, pB1, bias1, GX);
    lstm_rec<1><<<1, 1024, 0, stream>>>(nullptr, GX, pB1, bias1, nullptr, mh1,
                                        ht1, ct1);
  } else {
    lstm_rec<0><<<1, 1024, 0, stream>>>(X, nullptr, pB0, bias0, mi0, mh0,
                                        ht0, ct0);
    lstm_rec<0><<<1, 1024, 0, stream>>>(ht0, nullptr, pB1, bias1, mi1, mh1,
                                        ht1, ct1);
  }
}
